// AttentionBase_8040178778689
// MI455X (gfx1250) — matrix operand order repaired and checked by reading, not compiled, not run on hardware
//
#include <hip/hip_runtime.h>
#include <hip/hip_fp16.h>

typedef __attribute__((ext_vector_type(16))) _Float16 v16h;
typedef __attribute__((ext_vector_type(8)))  float    v8f;

namespace {
constexpr int Bb  = 4;
constexpr int Nn  = 2048;
constexpr int Hh  = 16;
constexpr int Dd  = 64;
constexpr int MID = Hh * Dd;   // 1024
}

// 16-lane (DPP16 row) rotate-reduce: after ror 1,2,4,8 every lane in the row
// holds the full 16-lane reduction. Pure VALU, no LDS traffic.
__device__ __forceinline__ float row16_max(float x) {
  x = fmaxf(x, __int_as_float(__builtin_amdgcn_update_dpp(0, __float_as_int(x), 0x121, 0xf, 0xf, true)));
  x = fmaxf(x, __int_as_float(__builtin_amdgcn_update_dpp(0, __float_as_int(x), 0x122, 0xf, 0xf, true)));
  x = fmaxf(x, __int_as_float(__builtin_amdgcn_update_dpp(0, __float_as_int(x), 0x124, 0xf, 0xf, true)));
  x = fmaxf(x, __int_as_float(__builtin_amdgcn_update_dpp(0, __float_as_int(x), 0x128, 0xf, 0xf, true)));
  return x;
}
__device__ __forceinline__ float row16_sum(float x) {
  x += __int_as_float(__builtin_amdgcn_update_dpp(0, __float_as_int(x), 0x121, 0xf, 0xf, true));
  x += __int_as_float(__builtin_amdgcn_update_dpp(0, __float_as_int(x), 0x122, 0xf, 0xf, true));
  x += __int_as_float(__builtin_amdgcn_update_dpp(0, __float_as_int(x), 0x124, 0xf, 0xf, true));
  x += __int_as_float(__builtin_amdgcn_update_dpp(0, __float_as_int(x), 0x128, 0xf, 0xf, true));
  return x;
}

// -------------------------------------------------------------------------
// Kernel 1: RoPE(q,k) + fp32->f16 convert of q,k,v into [B][H][N][D] layout
// -------------------------------------------------------------------------
__global__ void __launch_bounds__(256)
rope_convert_kernel(const float* __restrict__ q, const float* __restrict__ k,
                    const float* __restrict__ v,
                    _Float16* __restrict__ qh, _Float16* __restrict__ kh,
                    _Float16* __restrict__ vh) {
  const int idx = blockIdx.x * 256 + threadIdx.x;        // B*H*N*(D/2)
  const int i   = idx & 31;                              // pair index 0..31
  const int n   = (idx >> 5) & (Nn - 1);
  const int bh  = idx / (32 * Nn);
  const int h   = bh & (Hh - 1);
  const int b   = bh / Hh;

  const size_t src = ((size_t)(b * Nn + n)) * MID + h * Dd + 2 * i;
  const size_t dst = ((size_t)bh * Nn + n) * Dd + 2 * i;

  // inv_freq = ROPE_BASE^(-2i/D); ln(10000) = 9.210340371976184
  const float inv = __expf(-((float)(2 * i) / (float)Dd) * 9.210340371976184f);
  const float ang = (float)n * inv;
  const float sn = __sinf(ang);
  const float cs = __cosf(ang);

  const float qe = q[src], qo = q[src + 1];
  qh[dst]     = (_Float16)(qe * cs - qo * sn);
  qh[dst + 1] = (_Float16)(qe * sn + qo * cs);

  const float ke = k[src], ko = k[src + 1];
  kh[dst]     = (_Float16)(ke * cs - ko * sn);
  kh[dst + 1] = (_Float16)(ke * sn + ko * cs);

  vh[dst]     = (_Float16)v[src];
  vh[dst + 1] = (_Float16)v[src + 1];
}

// -------------------------------------------------------------------------
// Kernel 1b: convert w_out fp32 -> f16
// -------------------------------------------------------------------------
__global__ void __launch_bounds__(256)
wconv_kernel(const float* __restrict__ w, _Float16* __restrict__ w16) {
  const int idx = blockIdx.x * 256 + threadIdx.x;
  w16[idx] = (_Float16)w[idx];
}

// -------------------------------------------------------------------------
// Kernel 2: flash attention. 256 threads = 8 wave32, all on one (b,h).
// K/V 32x64 tiles are DMA'd into LDS with async global->LDS loads
// (double-buffered, ASYNCcnt), shared by all 8 waves. Each wave owns a
// 16-row Q tile: 4 WMMAs for S + 4 WMMAs for P*V per 32-column step.
// -------------------------------------------------------------------------
__global__ void __launch_bounds__(256)
flash_attn_kernel(const _Float16* __restrict__ qh, const _Float16* __restrict__ kh,
                  const _Float16* __restrict__ vh, _Float16* __restrict__ xo) {
  __shared__ __attribute__((aligned(32))) _Float16 kbuf[2][32 * 64];  // 8 KB
  __shared__ __attribute__((aligned(32))) _Float16 vbuf[2][32 * 64];  // 8 KB
  __shared__ __attribute__((aligned(32))) _Float16 pbuf[8][16 * 32];  // 8 KB

  const int t    = threadIdx.x;
  const int lane = t & 31;
  const int wave = t >> 5;                               // 0..7
  const int bh   = blockIdx.x >> 4;                      // 16 blocks per (b,h)
  const int qt   = (blockIdx.x & 15) * 8 + wave;         // Q tile index 0..127
  const int h    = bh & (Hh - 1);
  const int b    = bh / Hh;

  const size_t base = (size_t)bh * Nn * Dd;
  const int row  = lane & 15;
  const int kgrp = lane >> 4;                            // 0 or 1

  // --- per-thread async-DMA staging addresses (16 B per thread per tile) ---
  const int srow = t >> 3;                               // 0..31 tile row
  const int scol = (t & 7) * 8;                          // halves within row
  const unsigned k_lds0 = (unsigned)(uintptr_t)&kbuf[0][srow * 64 + scol];
  const unsigned k_lds1 = (unsigned)(uintptr_t)&kbuf[1][srow * 64 + scol];
  const unsigned v_lds0 = (unsigned)(uintptr_t)&vbuf[0][srow * 64 + scol];
  const unsigned v_lds1 = (unsigned)(uintptr_t)&vbuf[1][srow * 64 + scol];
  const _Float16* kgp = kh + base + (size_t)srow * Dd + scol;
  const _Float16* vgp = vh + base + (size_t)srow * Dd + scol;

  auto stage = [&](int buf, int m) {
    const unsigned kl = buf ? k_lds1 : k_lds0;
    const unsigned vl = buf ? v_lds1 : v_lds0;
    const unsigned long long ka = (unsigned long long)(uintptr_t)(kgp + (size_t)m * Dd);
    const unsigned long long va = (unsigned long long)(uintptr_t)(vgp + (size_t)m * Dd);
    asm volatile("global_load_async_to_lds_b128 %0, %1, off" :: "v"(kl), "v"(ka) : "memory");
    asm volatile("global_load_async_to_lds_b128 %0, %1, off" :: "v"(vl), "v"(va) : "memory");
  };

  // --- Q A-matrix fragments (16x32 f16 each): K chunks [0,32) and [32,64) ---
  const _Float16* qp = qh + base + (size_t)(qt * 16 + row) * Dd + kgrp * 16;
  const v16h aq0 = *(const v16h*)(qp);
  const v16h aq1 = *(const v16h*)(qp + 32);

  v8f acc[4];                                            // 16x64 f32 output tile
  acc[0] = v8f{}; acc[1] = v8f{}; acc[2] = v8f{}; acc[3] = v8f{};
  float mrun[8], lpart[8];
#pragma unroll
  for (int r = 0; r < 8; ++r) { mrun[r] = -3.0e38f; lpart[r] = 0.0f; }

  const float scale = 0.125f;                            // 1/sqrt(64)
  _Float16* pl = &pbuf[wave][0];

  stage(0, 0);                                           // prologue DMA

  const int nsteps = Nn / 32;
  for (int mi = 0; mi < nsteps; ++mi) {
    const int buf = mi & 1;
    if (mi + 1 < nsteps) {
      stage(buf ^ 1, (mi + 1) * 32);                     // overlap next DMA
      asm volatile("s_wait_asynccnt 2" ::: "memory");    // oldest 2 (this tile) done
    } else {
      asm volatile("s_wait_asynccnt 0" ::: "memory");
    }
    __syncthreads();                                     // tile visible to all 8 waves

    const _Float16* kt = &kbuf[buf][0];
    const _Float16* vt = &vbuf[buf][0];

    // ---- issue all K fragment loads up front so they pipeline ----
    const _Float16* kr0 = kt + row * 64 + kgrp * 16;
    const _Float16* kr1 = kt + (16 + row) * 64 + kgrp * 16;
    const v16h bk00 = *(const v16h*)(kr0);
    const v16h bk01 = *(const v16h*)(kr0 + 32);
    const v16h bk10 = *(const v16h*)(kr1);
    const v16h bk11 = *(const v16h*)(kr1 + 32);

    // ---- scores S[16 x 32] = Q x K^T (two 16x16 column tiles) ----
    v8f s0 = v8f{}, s1 = v8f{};
    s0 = __builtin_amdgcn_wmma_f32_16x16x32_f16(false, aq0, false, bk00,
                                                (short)0, s0, false, false);
    s1 = __builtin_amdgcn_wmma_f32_16x16x32_f16(false, aq0, false, bk10,
                                                (short)0, s1, false, false);
    s0 = __builtin_amdgcn_wmma_f32_16x16x32_f16(false, aq1, false, bk01,
                                                (short)0, s0, false, false);
    s1 = __builtin_amdgcn_wmma_f32_16x16x32_f16(false, aq1, false, bk11,
                                                (short)0, s1, false, false);

    // ---- preload V fragments now: LDS latency hides under softmax VALU ----
    v16h bv0, bv1, bv2, bv3;
#pragma unroll
    for (int s = 0; s < 16; ++s) {
      const _Float16* vr = vt + (16 * kgrp + s) * 64 + row;
      bv0[s] = vr[0];
      bv1[s] = vr[16];
      bv2[s] = vr[32];
      bv3[s] = vr[48];
    }

    // ---- online softmax: DPP16 row reductions, lane-local deferred sums ----
#pragma unroll
    for (int r = 0; r < 8; ++r) {
      const float a = s0[r] * scale;
      const float c = s1[r] * scale;
      const float mx   = row16_max(fmaxf(a, c));
      const float mnew = fmaxf(mrun[r], mx);
      const float fac  = __expf(mrun[r] - mnew);
      mrun[r] = mnew;
      const float p0 = __expf(a - mnew);
      const float p1 = __expf(c - mnew);
      lpart[r] = lpart[r] * fac + p0 + p1;               // reduced once at the end
      acc[0][r] *= fac; acc[1][r] *= fac; acc[2][r] *= fac; acc[3][r] *= fac;
      // stage P (C-layout -> A-layout transpose through LDS)
      const int rowm = r + 8 * kgrp;
      pl[rowm * 32 + row]      = (_Float16)p0;
      pl[rowm * 32 + 16 + row] = (_Float16)p1;
    }

    // wave-local LDS visibility before cross-lane re-read (CDNA5 split counter)
    asm volatile("s_wait_dscnt 0" ::: "memory");

    const v16h pa = *(const v16h*)(pl + row * 32 + kgrp * 16);

    // ---- acc += P x V : 4 output column tiles of 16 ----
    acc[0] = __builtin_amdgcn_wmma_f32_16x16x32_f16(false, pa, false, bv0,
                                                    (short)0, acc[0], false, false);
    acc[1] = __builtin_amdgcn_wmma_f32_16x16x32_f16(false, pa, false, bv1,
                                                    (short)0, acc[1], false, false);
    acc[2] = __builtin_amdgcn_wmma_f32_16x16x32_f16(false, pa, false, bv2,
                                                    (short)0, acc[2], false, false);
    acc[3] = __builtin_amdgcn_wmma_f32_16x16x32_f16(false, pa, false, bv3,
                                                    (short)0, acc[3], false, false);

    __syncthreads();                                     // all reads done before overwrite
  }

  // ---- finalize: reduce row sums across the 16-lane row, normalize, store ----
#pragma unroll
  for (int r = 0; r < 8; ++r) {
    const float l = row16_sum(lpart[r]);
    const float inv = 1.0f / l;
    const int grow = qt * 16 + r + 8 * kgrp;
    _Float16* op = xo + ((size_t)(b * Nn + grow)) * MID + h * Dd + row;
#pragma unroll
    for (int dt = 0; dt < 4; ++dt) {
      op[dt * 16] = (_Float16)(acc[dt][r] * inv);
    }
  }
}

// -------------------------------------------------------------------------
// Kernel 3: out = x(f16) @ w_out^T(f16) + b_out, f32 output.
// One wave per 16x16 output tile; K=1024 in 32-steps (32 WMMAs).
// -------------------------------------------------------------------------
__global__ void __launch_bounds__(256)
proj_kernel(const _Float16* __restrict__ x, const _Float16* __restrict__ w16,
            const float* __restrict__ bias, float* __restrict__ out) {
  const int lane = threadIdx.x & 31;
  const int wave = threadIdx.x >> 5;
  const int gw   = blockIdx.x * 8 + wave;                // (B*N/16)*(MID/16) waves
  const int jtiles = MID / 16;                           // 64
  const int jt = gw % jtiles;
  const int it = gw / jtiles;

  const int row  = lane & 15;
  const int kgrp = lane >> 4;

  const _Float16* xp = x   + (size_t)(it * 16 + row) * MID + kgrp * 16;
  const _Float16* wp = w16 + (size_t)(jt * 16 + row) * MID + kgrp * 16;

  v8f acc = v8f{};
#pragma unroll 4
  for (int k = 0; k < MID; k += 32) {
    const v16h a = *(const v16h*)(xp + k);
    const v16h b = *(const v16h*)(wp + k);
    acc = __builtin_amdgcn_wmma_f32_16x16x32_f16(false, a, false, b,
                                                 (short)0, acc, false, false);
  }

  const float bj = bias[jt * 16 + row];                  // lane indexes output column
#pragma unroll
  for (int r = 0; r < 8; ++r) {
    const int orow = it * 16 + r + 8 * kgrp;
    out[(size_t)orow * MID + jt * 16 + row] = acc[r] + bj;
  }
}

// -------------------------------------------------------------------------
extern "C" void kernel_launch(void* const* d_in, const int* in_sizes, int n_in,
                              void* d_out, int out_size, void* d_ws, size_t ws_size,
                              hipStream_t stream) {
  (void)in_sizes; (void)n_in; (void)out_size; (void)ws_size;

  const float* q     = (const float*)d_in[0];
  const float* k     = (const float*)d_in[1];
  const float* v     = (const float*)d_in[2];
  const float* w_out = (const float*)d_in[3];
  const float* b_out = (const float*)d_in[4];
  float* out = (float*)d_out;

  // workspace layout (f16 elements)
  const size_t QKV = (size_t)Bb * Hh * Nn * Dd;          // 8,388,608
  const size_t X   = (size_t)Bb * Nn * MID;              // 8,388,608
  _Float16* qh  = (_Float16*)d_ws;
  _Float16* kh  = qh + QKV;
  _Float16* vh  = kh + QKV;
  _Float16* x16 = vh + QKV;
  _Float16* w16 = x16 + X;                               // total ~66 MB f16

  // 1) RoPE + convert: B*H*N*(D/2) threads
  {
    const int total = Bb * Hh * Nn * (Dd / 2);           // 4,194,304
    rope_convert_kernel<<<total / 256, 256, 0, stream>>>(q, k, v, qh, kh, vh);
  }
  // 1b) weight convert
  wconv_kernel<<<(MID * MID) / 256, 256, 0, stream>>>(w_out, w16);

  // 2) flash attention: 8 waves (8 Q tiles) per block, 16 blocks per (b,h)
  {
    const int blocks = Bb * Hh * (Nn / (16 * 8));        // 1024
    flash_attn_kernel<<<blocks, 256, 0, stream>>>(qh, kh, vh, x16);
  }

  // 3) projection GEMM: (B*N/16)*(MID/16) = 32768 waves, 8 per block
  {
    const int waves = (Bb * Nn / 16) * (MID / 16);
    proj_kernel<<<waves / 8, 256, 0, stream>>>(x16, w16, b_out, out);
  }
}
